// CosSimRouter_pad_merge_entropy_49778670960800
// MI455X (gfx1250) — matrix operand order repaired and checked by reading, not compiled, not run on hardware
//
#include <hip/hip_runtime.h>
#include <hip/hip_bf16.h>

// Problem constants (match reference)
#define LV   1024
#define LT   128
#define DD   4096
#define NH   8
#define DHD  512
#define HIDN 2048
#define NK1  512
#define NK2  256
#define NREM 768

typedef __attribute__((ext_vector_type(16))) __bf16 v16bf;
typedef __attribute__((ext_vector_type(4)))  __bf16 v4bf;
typedef __attribute__((ext_vector_type(8)))  float  v8f;
typedef __attribute__((ext_vector_type(4)))  float  f32x4;
typedef __attribute__((ext_vector_type(4)))  unsigned int u32x4;

union FragBF { v16bf v; u32x4 u[2]; };
union Pack8  { __bf16 e[8]; u32x4 q; };

// ---------------------------------------------------------------------------
// WMMA GEMM:  C[m,n] = alpha * sum_k A[m,k]*B[n,k] + bias[n]
//
// Block tile: 128 (M) x 64 (N); 8 waves, wave w owns rows [tileM, tileM+16)
// and all 64 columns (4 accumulators).  The 64x32 B panel is staged in LDS
// with DOUBLE BUFFERING: while the WMMAs consume panel k, the 256 threads
// stage panel k+1 (fp32 weights converted to bf16 in-register during the
// fill) -> one workgroup barrier per K-step and the global weight stream is
// hidden behind the matrix math.  All four B fragments are pulled into
// distinct registers before the WMMA burst so the ds_loads can overlap.
//
// REQUIRES: M % 128 == 0, N % 64 == 0, K % 32 == 0 (all call sites comply).
// Batched over blockIdx.z with element strides sA/sB/sC (attention heads).
// ---------------------------------------------------------------------------
template<bool B_IS_F32, bool RELU>
__launch_bounds__(256)
__global__ void wmma_gemm_bt(const __bf16* __restrict__ A, const void* __restrict__ Braw,
                             const float* __restrict__ bias,
                             float* __restrict__ C, __bf16* __restrict__ Cbf,
                             int M, int N, int K, int lda, int ldb, int ldc,
                             long long sA, long long sB, long long sC, float alpha)
{
    __shared__ __bf16 Bs[2][64 * 32];       // two 64x32 panels, 2 x 4 KB

    const int lane  = threadIdx.x & 31;
    const int wave  = threadIdx.x >> 5;
    const int tileM = blockIdx.y * 128 + wave * 16;
    const int tileN = blockIdx.x * 64;

    const long long zA = (long long)blockIdx.z * sA;
    const long long zB = (long long)blockIdx.z * sB;
    const long long zC = (long long)blockIdx.z * sC;

    // A fragment: lane<16 holds row M=lane,    K = {0..7, 16..23}+k0
    //             lane>=16 holds row M=lane-16, K = {8..15, 24..31}+k0
    const int arow  = tileM + (lane & 15);
    const int akoff = (lane < 16) ? 0 : 8;
    const __bf16* aBase = A + zA + (long long)arow * lda + akoff;

    // LDS fill assignment: thread -> (column, 8-wide K chunk)
    const int fcol = threadIdx.x >> 2;        // 0..63
    const int fk   = (threadIdx.x & 3) * 8;   // 0, 8, 16, 24
    const long long fBase = zB + (long long)(tileN + fcol) * ldb + fk;

    // B fragment read: lane<16 -> local col = t*16+lane,    K = {0..7, 16..23}
    //                  lane>=16 -> local col = t*16+lane-16, K = {8..15, 24..31}
    const int bl  = lane & 15;
    const int bko = (lane < 16) ? 0 : 8;

    // stage one 64x32 panel for K-base k0 into buffer `buf`
    auto stage = [&](int k0, int buf) {
        if constexpr (B_IS_F32) {
            const float* bp = (const float*)Braw + fBase + k0;
            f32x4 x0 = reinterpret_cast<const f32x4*>(bp)[0];
            f32x4 x1 = reinterpret_cast<const f32x4*>(bp)[1];
            Pack8 pk;
            #pragma unroll
            for (int i = 0; i < 4; ++i) { pk.e[i] = (__bf16)x0[i]; pk.e[4 + i] = (__bf16)x1[i]; }
            *reinterpret_cast<u32x4*>(&Bs[buf][fcol * 32 + fk]) = pk.q;
        } else {
            const __bf16* bp = (const __bf16*)Braw + fBase + k0;
            *reinterpret_cast<u32x4*>(&Bs[buf][fcol * 32 + fk]) = *reinterpret_cast<const u32x4*>(bp);
        }
    };

    v8f acc[4];
    #pragma unroll
    for (int t = 0; t < 4; ++t)
        #pragma unroll
        for (int i = 0; i < 8; ++i) acc[t][i] = 0.0f;

    const int nsteps = K >> 5;
    stage(0, 0);
    __syncthreads();

    for (int it = 0; it < nsteps; ++it) {
        const int cur = it & 1;
        if (it + 1 < nsteps) stage((it + 1) << 5, cur ^ 1);   // prefetch next panel

        FragBF fa;
        fa.u[0] = *reinterpret_cast<const u32x4*>(aBase + (it << 5));
        fa.u[1] = *reinterpret_cast<const u32x4*>(aBase + (it << 5) + 16);

        FragBF fb[4];
        #pragma unroll
        for (int t = 0; t < 4; ++t) {
            const __bf16* ls = &Bs[cur][(t * 16 + bl) * 32 + bko];
            fb[t].u[0] = reinterpret_cast<const u32x4*>(ls)[0];
            fb[t].u[1] = reinterpret_cast<const u32x4*>(ls)[2];
        }
        #pragma unroll
        for (int t = 0; t < 4; ++t)
            acc[t] = __builtin_amdgcn_wmma_f32_16x16x32_bf16(false, fa.v, false, fb[t].v,
                                                             (short)0, acc[t], false, false);
        __syncthreads();
    }

    // C/D layout: lane<16 -> n=col, rows tileM+0..7; lane>=16 -> rows tileM+8..15
    const int row0 = tileM + ((lane >> 4) << 3);
    #pragma unroll
    for (int t = 0; t < 4; ++t) {
        const int col = tileN + t * 16 + bl;
        const float bv = bias ? bias[col] : 0.0f;
        #pragma unroll
        for (int r = 0; r < 8; ++r) {
            float val = acc[t][r] * alpha + bv;
            if (RELU) val = fmaxf(val, 0.0f);
            const long long ci = zC + (long long)(row0 + r) * ldc + col;
            if (C)   C[ci]   = val;
            if (Cbf) Cbf[ci] = (__bf16)val;
        }
    }
}

// ---------------------------------------------------------------------------
__global__ void cvt_f32_to_bf16(const float* __restrict__ in, __bf16* __restrict__ out,
                                long long n)
{
    long long i = ((long long)blockIdx.x * 256 + threadIdx.x) * 4;
    if (i + 3 < n) {
        f32x4 x = *reinterpret_cast<const f32x4*>(in + i);
        v4bf y;
        y[0] = (__bf16)x[0]; y[1] = (__bf16)x[1]; y[2] = (__bf16)x[2]; y[3] = (__bf16)x[3];
        *reinterpret_cast<v4bf*>(out + i) = y;
    }
}

// Transpose bf16 (R x C) -> (C x R) via LDS tile (for per-head V^T).
__global__ void transpose_bf16(const __bf16* __restrict__ in, __bf16* __restrict__ out,
                               int R, int C)
{
    __shared__ __bf16 tile[32][33];
    const int c0 = blockIdx.x * 32;
    const int r0 = blockIdx.y * 32;
    for (int i = threadIdx.y; i < 32; i += 8)
        tile[i][threadIdx.x] = in[(long long)(r0 + i) * C + (c0 + threadIdx.x)];
    __syncthreads();
    for (int i = threadIdx.y; i < 32; i += 8)
        out[(long long)(c0 + i) * R + (r0 + threadIdx.x)] = tile[threadIdx.x][i];
}

// Row softmax (f32 in, bf16 out). One 256-thread block per row; R <= 2048.
__global__ void softmax_rows(const float* __restrict__ S, __bf16* __restrict__ P, int R)
{
    __shared__ float red[256];
    const long long base = (long long)blockIdx.x * R;
    const int t = threadIdx.x;
    float e[8];
    float mx = -3.4e38f;
    for (int j = t; j < R; j += 256) mx = fmaxf(mx, S[base + j]);
    red[t] = mx; __syncthreads();
    for (int s = 128; s > 0; s >>= 1) { if (t < s) red[t] = fmaxf(red[t], red[t + s]); __syncthreads(); }
    mx = red[0];
    __syncthreads();
    float sum = 0.0f; int c = 0;
    for (int j = t; j < R; j += 256, ++c) { float v = expf(S[base + j] - mx); e[c] = v; sum += v; }
    red[t] = sum; __syncthreads();
    for (int s = 128; s > 0; s >>= 1) { if (t < s) red[t] += red[t + s]; __syncthreads(); }
    const float inv = 1.0f / red[0];
    c = 0;
    for (int j = t; j < R; j += 256, ++c) P[base + j] = (__bf16)(e[c] * inv);
}

// logits[row] = dot(H[row, 0..K-1], w2) + b2[0]
__global__ void gemv_logits(const float* __restrict__ Hm, const float* __restrict__ w2,
                            const float* __restrict__ b2, float* __restrict__ out, int K)
{
    __shared__ float red[256];
    const int row = blockIdx.x, t = threadIdx.x;
    const float* hp = Hm + (long long)row * K;
    float s = 0.0f;
    for (int j = t; j < K; j += 256) s += hp[j] * w2[j];
    red[t] = s; __syncthreads();
    for (int st = 128; st > 0; st >>= 1) { if (t < st) red[t] += red[t + st]; __syncthreads(); }
    if (t == 0) out[row] = red[0] + b2[0];
}

// Full bitonic sort of n (power of two) (value,index) pairs, descending value,
// ties -> lower index first (matches jax.lax.top_k). blockDim.x == n.
__global__ void topk_sort(const float* __restrict__ vals, int n, int keep, int* __restrict__ out)
{
    __shared__ float sv[1024];
    __shared__ int   si[1024];
    const int t = threadIdx.x;
    if (t < n) { sv[t] = vals[t]; si[t] = t; }
    __syncthreads();
    for (int k = 2; k <= n; k <<= 1) {
        for (int j = k >> 1; j > 0; j >>= 1) {
            const int p = t ^ j;
            if (t < n && p > t) {
                const bool up = ((t & k) == 0);
                float av = sv[t], bv = sv[p];
                int   ai = si[t], bi = si[p];
                // "a belongs after b" in descending-with-index-tiebreak order
                const bool aAfter = (av < bv) || (av == bv && ai > bi);
                if (up == aAfter) { sv[t] = bv; si[t] = bi; sv[p] = av; si[p] = ai; }
            }
            __syncthreads();
        }
    }
    if (t < keep) out[t] = si[t];
}

// dst[r,:] = src[idx[r],:]   (bf16 rows of width DD, 128-bit copies)
__global__ void gather_rows_bf16(const __bf16* __restrict__ src, const int* __restrict__ idx,
                                 __bf16* __restrict__ dst)
{
    const int r = blockIdx.x;
    const u32x4* s = reinterpret_cast<const u32x4*>(src + (long long)idx[r] * DD);
    u32x4*       d = reinterpret_cast<u32x4*>(dst + (long long)r * DD);
    for (int j = threadIdx.x; j < DD / 8; j += 256) d[j] = s[j];
}

__global__ void compose_idx(const int* __restrict__ idx1, const int* __restrict__ idx2,
                            int* __restrict__ selidx)
{
    const int t = threadIdx.x;
    if (t < NK2) selidx[t] = idx1[idx2[t]];
}

// rem_idx = increasing complement of sel_idx in [0, LV)
__global__ void complement_kernel(const int* __restrict__ selidx, int* __restrict__ remidx)
{
    __shared__ unsigned char mask[LV];
    const int t = threadIdx.x;
    for (int i = t; i < LV; i += 256) mask[i] = 0;
    __syncthreads();
    if (t < NK2) mask[selidx[t]] = 1;
    __syncthreads();
    if (t == 0) {
        int c = 0;
        for (int i = 0; i < LV; ++i) if (!mask[i]) remidx[c++] = i;
    }
}

// out[r] = || vf[idx[r], :] ||_2
__global__ void row_norm_idx(const float* __restrict__ vf, const int* __restrict__ idx,
                             float* __restrict__ out)
{
    __shared__ float red[256];
    const int r = blockIdx.x, t = threadIdx.x;
    const float* p = vf + (long long)idx[r] * DD;
    float s = 0.0f;
    for (int j = t; j < DD; j += 256) { float x = p[j]; s += x * x; }
    red[t] = s; __syncthreads();
    for (int st = 128; st > 0; st >>= 1) { if (t < st) red[t] += red[t + st]; __syncthreads(); }
    if (t == 0) out[r] = sqrtf(red[0]);
}

// best[i] = argmax_j dot(i,j) / max(nr[i]*ns[j], 1e-8); ties -> lowest j.
__global__ void argmax_rows(const float* __restrict__ sim, const float* __restrict__ nr,
                            const float* __restrict__ ns, int* __restrict__ best)
{
    __shared__ float bv[256];
    __shared__ int   bi[256];
    const int i = blockIdx.x, t = threadIdx.x;
    const float nri = nr[i];
    bv[t] = sim[(long long)i * NK2 + t] / fmaxf(nri * ns[t], 1e-8f);
    bi[t] = t;
    __syncthreads();
    for (int s = 128; s > 0; s >>= 1) {
        if (t < s) {
            if (bv[t + s] > bv[t] || (bv[t + s] == bv[t] && bi[t + s] < bi[t])) {
                bv[t] = bv[t + s]; bi[t] = bi[t + s];
            }
        }
        __syncthreads();
    }
    if (t == 0) best[i] = bi[0];
}

// Deterministic per-cluster merge: mean = (sel + sum(rem in cluster)) / (cnt+1),
// out = cnt>0 ? mean/||mean|| * max(ns, segmax(nr)) : sel
__global__ void finalize_merge(const float* __restrict__ vf, const int* __restrict__ selidx,
                               const int* __restrict__ remidx, const int* __restrict__ best,
                               const float* __restrict__ nr, const float* __restrict__ ns,
                               float* __restrict__ out)
{
    __shared__ float red[256];
    __shared__ int   bS[NREM];
    __shared__ int   rS[NREM];
    __shared__ float nS[NREM];
    const int c = blockIdx.x, t = threadIdx.x;
    for (int i = t; i < NREM; i += 256) { bS[i] = best[i]; rS[i] = remidx[i]; nS[i] = nr[i]; }
    __syncthreads();

    const float* selrow = vf + (long long)selidx[c] * DD;
    float m[DD / 256];
    #pragma unroll
    for (int u = 0; u < DD / 256; ++u) m[u] = selrow[t + u * 256];

    float cnt = 0.0f, smax = 0.0f;
    for (int i = 0; i < NREM; ++i) {
        if (bS[i] == c) {
            cnt += 1.0f;
            smax = fmaxf(smax, nS[i]);
            const float* rr = vf + (long long)rS[i] * DD;
            #pragma unroll
            for (int u = 0; u < DD / 256; ++u) m[u] += rr[t + u * 256];
        }
    }
    const float inv = 1.0f / (cnt + 1.0f);
    float ss = 0.0f;
    #pragma unroll
    for (int u = 0; u < DD / 256; ++u) { m[u] *= inv; ss += m[u] * m[u]; }
    red[t] = ss; __syncthreads();
    for (int s = 128; s > 0; s >>= 1) { if (t < s) red[t] += red[t + s]; __syncthreads(); }
    const float nm = sqrtf(red[0]);
    const float mx = fmaxf(ns[c], smax);           // smax==0 when cnt==0
    const bool merged = (cnt > 0.0f);
    const float sc = merged ? (mx / nm) : 0.0f;
    #pragma unroll
    for (int u = 0; u < DD / 256; ++u) {
        const int d = t + u * 256;
        out[(long long)c * DD + d] = merged ? (m[u] * sc) : selrow[d];
    }
}

// ---------------------------------------------------------------------------
static inline void gemm(hipStream_t st, const __bf16* A, const void* B, bool bF32,
                        const float* bias, float* C, __bf16* Cbf, bool relu,
                        int M, int N, int K, int lda, int ldb, int ldc,
                        long long sA, long long sB, long long sC, float alpha, int batch)
{
    dim3 grid((unsigned)(N / 64), (unsigned)(M / 128), (unsigned)batch), block(256);
    if (bF32) {
        if (relu)
            wmma_gemm_bt<true,  true ><<<grid, block, 0, st>>>(A, B, bias, C, Cbf, M, N, K, lda, ldb, ldc, sA, sB, sC, alpha);
        else
            wmma_gemm_bt<true,  false><<<grid, block, 0, st>>>(A, B, bias, C, Cbf, M, N, K, lda, ldb, ldc, sA, sB, sC, alpha);
    } else {
        wmma_gemm_bt<false, false><<<grid, block, 0, st>>>(A, B, bias, C, Cbf, M, N, K, lda, ldb, ldc, sA, sB, sC, alpha);
    }
}

extern "C" void kernel_launch(void* const* d_in, const int* in_sizes, int n_in,
                              void* d_out, int out_size, void* d_ws, size_t ws_size,
                              hipStream_t stream)
{
    (void)in_sizes; (void)n_in; (void)out_size; (void)ws_size;

    const float* vf        = (const float*)d_in[0];
    const float* te        = (const float*)d_in[1];
    /* d_in[2] attention_mask: unused by reference */
    const float* attn_in_w = (const float*)d_in[3];
    const float* attn_in_b = (const float*)d_in[4];
    const float* attn_out_w= (const float*)d_in[5];
    const float* attn_out_b= (const float*)d_in[6];
    const float* mlp_w1    = (const float*)d_in[7];
    const float* mlp_b1    = (const float*)d_in[8];
    const float* mlp_w2    = (const float*)d_in[9];
    const float* mlp_b2    = (const float*)d_in[10];

    // ---- workspace arena (256B aligned) ----
    char* w = (char*)d_ws;
    auto take = [&](size_t bytes) { char* p = w; w += (bytes + 255) & ~(size_t)255; return p; };
    __bf16* comb1  = (__bf16*)take((size_t)(LV + LT) * DD * 2);     // [vf ; te] bf16
    __bf16* comb2  = (__bf16*)take((size_t)(NK1 + LT) * DD * 2);    // [sv1 ; te] bf16
    __bf16* qbf    = (__bf16*)take((size_t)LV * DD * 2);
    __bf16* kbf    = (__bf16*)take((size_t)(LV + LT) * DD * 2);
    __bf16* vbf    = (__bf16*)take((size_t)(LV + LT) * DD * 2);
    __bf16* vT     = (__bf16*)take((size_t)DD * (LV + LT) * 2);     // per-head transposed V
    float*  scores = (float*) take((size_t)NH * LV * (LV + LT) * 4);
    __bf16* probs  = (__bf16*)take((size_t)NH * LV * (LV + LT) * 2);
    __bf16* obf    = (__bf16*)take((size_t)LV * DD * 2);
    __bf16* gbf    = (__bf16*)take((size_t)LV * DD * 2);
    float*  hbuf   = (float*) take((size_t)LV * HIDN * 4);
    float*  logits = (float*) take((size_t)LV * 4);
    int*    idx1   = (int*)   take(NK1 * 4);
    int*    idx2   = (int*)   take(NK2 * 4);
    int*    selidx = (int*)   take(NK2 * 4);
    int*    remidx = (int*)   take(NREM * 4);
    int*    best   = (int*)   take(NREM * 4);
    float*  nr     = (float*) take(NREM * 4);
    float*  ns     = (float*) take(NK2 * 4);
    __bf16* selbf  = (__bf16*)take((size_t)NK2 * DD * 2);
    __bf16* rembf  = (__bf16*)take((size_t)NREM * DD * 2);
    float*  sim    = (float*) take((size_t)NREM * NK2 * 4);

    const float inv_sqrt_dh = 0.044194173824159216f; // 1/sqrt(512)

    // ---- activations -> bf16 (comb1 = [vf ; te]) ----
    {
        long long n1 = (long long)LV * DD, n2 = (long long)LT * DD;
        cvt_f32_to_bf16<<<(unsigned)(n1 / 1024), 256, 0, stream>>>(vf, comb1, n1);
        cvt_f32_to_bf16<<<(unsigned)(n2 / 1024), 256, 0, stream>>>(te, comb1 + n1, n2);
    }

    // ---- one selection module (global branch only; local branch is dead code) ----
    auto run_module = [&](const __bf16* Ain, int Lq, int Lkv,
                          const float* inw, const float* inb,
                          const float* outw, const float* outb,
                          const float* w1, const float* b1,
                          const float* w2, const float* b2,
                          int keep, int* idxout) {
        const size_t DxD = (size_t)DD * DD;
        // Q/K/V projections (weights fp32, converted in-register during LDS fill)
        gemm(stream, Ain, inw,            true, inb,          nullptr, qbf, false,
             Lq,  DD, DD, DD, DD, DD, 0, 0, 0, 1.0f, 1);
        gemm(stream, Ain, inw + DxD,      true, inb + DD,     nullptr, kbf, false,
             Lkv, DD, DD, DD, DD, DD, 0, 0, 0, 1.0f, 1);
        gemm(stream, Ain, inw + 2 * DxD,  true, inb + 2 * DD, nullptr, vbf, false,
             Lkv, DD, DD, DD, DD, DD, 0, 0, 0, 1.0f, 1);
        // V^T (per head rows become contiguous-K for the B^T GEMM path)
        transpose_bf16<<<dim3(DD / 32, Lkv / 32), dim3(32, 8), 0, stream>>>(vbf, vT, Lkv, DD);
        // scores[h] = Q_h K_h^T / sqrt(dh)
        gemm(stream, qbf, kbf, false, nullptr, scores, nullptr, false,
             Lq, Lkv, DHD, DD, DD, Lkv,
             (long long)DHD, (long long)DHD, (long long)Lq * Lkv, inv_sqrt_dh, NH);
        softmax_rows<<<NH * Lq, 256, 0, stream>>>(scores, probs, Lkv);
        // O[:, h*dh : (h+1)*dh] = P_h V_h
        gemm(stream, probs, vT, false, nullptr, nullptr, obf, false,
             Lq, DHD, Lkv, Lkv, Lkv, DD,
             (long long)Lq * Lkv, (long long)DHD * Lkv, (long long)DHD, 1.0f, NH);
        // output projection
        gemm(stream, obf, outw, true, outb, nullptr, gbf, false,
             Lq, DD, DD, DD, DD, DD, 0, 0, 0, 1.0f, 1);
        // MLP hidden (global channel only) with fused ReLU
        gemm(stream, gbf, w1, true, b1, hbuf, nullptr, true,
             Lq, HIDN, DD, DD, DD, HIDN, 0, 0, 0, 1.0f, 1);
        gemv_logits<<<Lq, 256, 0, stream>>>(hbuf, w2, b2, logits, HIDN);
        topk_sort<<<1, Lq, 0, stream>>>(logits, Lq, keep, idxout);
    };

    const size_t W3 = (size_t)3 * DD * DD;
    // module 1: params index 0 (global), mlp index 0
    run_module(comb1, LV, LV + LT,
               attn_in_w, attn_in_b, attn_out_w, attn_out_b,
               mlp_w1, mlp_b1, mlp_w2, mlp_b2, NK1, idx1);

    // comb2 = [vf[idx1] ; te]  (bf16)
    gather_rows_bf16<<<NK1, 256, 0, stream>>>(comb1, idx1, comb2);
    hipMemcpyAsync(comb2 + (size_t)NK1 * DD, comb1 + (size_t)LV * DD,
                   (size_t)LT * DD * sizeof(__bf16), hipMemcpyDeviceToDevice, stream);

    // module 2: params index 2 (global), mlp index 1
    run_module(comb2, NK1, NK1 + LT,
               attn_in_w + 2 * W3, attn_in_b + 2 * (3 * DD),
               attn_out_w + 2 * (size_t)DD * DD, attn_out_b + 2 * DD,
               mlp_w1 + (size_t)HIDN * DD, mlp_b1 + HIDN,
               mlp_w2 + HIDN, mlp_b2 + 1, NK2, idx2);

    // ---- merge ----
    compose_idx<<<1, 256, 0, stream>>>(idx1, idx2, selidx);
    complement_kernel<<<1, 256, 0, stream>>>(selidx, remidx);
    row_norm_idx<<<NK2, 256, 0, stream>>>(vf, selidx, ns);
    row_norm_idx<<<NREM, 256, 0, stream>>>(vf, remidx, nr);
    gather_rows_bf16<<<NK2, 256, 0, stream>>>(comb1, selidx, selbf);
    gather_rows_bf16<<<NREM, 256, 0, stream>>>(comb1, remidx, rembf);
    // rem @ sel^T (cosine numerator)
    gemm(stream, rembf, selbf, false, nullptr, sim, nullptr, false,
         NREM, NK2, DD, DD, DD, NK2, 0, 0, 0, 1.0f, 1);
    argmax_rows<<<NREM, 256, 0, stream>>>(sim, nr, ns, best);
    finalize_merge<<<NK2, 256, 0, stream>>>(vf, selidx, remidx, best, nr, ns, (float*)d_out);
}
